// TransformerBlock_14516989460854
// MI455X (gfx1250) — matrix operand order repaired and checked by reading, not compiled, not run on hardware
//
#include <hip/hip_runtime.h>
#include <hip/hip_bf16.h>
#include <stdint.h>

typedef __attribute__((ext_vector_type(16))) __bf16 v16bf;
typedef __attribute__((ext_vector_type(8)))  float  v8f;
typedef unsigned int v4u __attribute__((ext_vector_type(4)));
typedef unsigned int v8u __attribute__((ext_vector_type(8)));

// ---------------------------------------------------------------- WMMA helper
static __device__ __forceinline__ v8f wmma_bf16(v16bf a, v16bf b, v8f c) {
  // (neg_a, A, neg_b, B, c_mod, C, reuse_a, reuse_b)
  return __builtin_amdgcn_wmma_f32_16x16x32_bf16(false, a, false, b, (short)0, c,
                                                 false, false);
}

union FragU { uint4 u[2]; v16bf v; };

// A-matrix 16x32 bf16 fragment (rows row0..row0+15, k = k0..k0+31), row-major src.
// Per ISA layout: lanes 0-15 hold K=k0+0..7 and k0+16..23; lanes 16-31 hold
// K=k0+8..15 and k0+24..31 for row = lane%16.  => two contiguous 16B loads.
static __device__ __forceinline__ v16bf load_a_g(const __hip_bfloat16* __restrict__ src,
                                                 int ld, int row0, int k0) {
  int lane = threadIdx.x & 31, r = lane & 15, half = lane >> 4;
  const char* p = (const char*)(src + (size_t)(row0 + r) * ld + k0 + half * 8);
  FragU f;
  f.u[0] = *(const uint4*)p;
  f.u[1] = *(const uint4*)(p + 32);
  return f.v;
}

// Same A fragment but from an LDS tile with byte row-stride strideB;
// base points at row 0 of the 16-row tile.
static __device__ __forceinline__ v16bf load_a_lds(const char* base, int strideB, int k0) {
  int lane = threadIdx.x & 31, r = lane & 15, half = lane >> 4;
  const char* p = base + r * strideB + (k0 + half * 8) * 2;
  FragU f;
  f.u[0] = *(const uint4*)p;
  f.u[1] = *(const uint4*)(p + 32);
  return f.v;
}

// B-matrix 32x16 bf16 fragment: column n = col0 + lane%16, K = k0..k0+31.
// lanes 0-15 hold K=k0+0..7 and k0+16..23; lanes 16-31 hold K=k0+8..15 and k0+24..31.
// B[k][n] = src[col0+n][k]  (src row-major, i.e. "W^T GEMM" operand).
static __device__ __forceinline__ v16bf load_b_g(const __hip_bfloat16* __restrict__ src,
                                                 int ld, int col0, int k0) {
  int lane = threadIdx.x & 31, r = lane & 15, half = lane >> 4;
  const char* p = (const char*)(src + (size_t)(col0 + r) * ld + k0 + half * 8);
  FragU f;
  f.u[0] = *(const uint4*)p;
  f.u[1] = *(const uint4*)(p + 32);
  return f.v;
}

// ---------------------------------------------------------------- fp32 -> bf16
__global__ __launch_bounds__(256) void tb_cvt_bf16(const float* __restrict__ src,
                                                   __hip_bfloat16* __restrict__ dst, int n) {
  int i = blockIdx.x * 256 + threadIdx.x;
  if (i < n) dst[i] = __float2bfloat16(src[i]);
}

// ---------------------------------------------------------------- GEMM
// C[row, col] = sum_k A[row,k] * W[col,k]   (M=8192, N=1024, K=1024)
// mode 0/1: q/k -> bf16 head-split layout [b,h,n,dh]; split order: h = col%16, dh = col/16
// mode 2  : v   -> fp32 plain [row, col]
// mode 3  : out -> fp32 plain [row, col] + bias[col]
__global__ __launch_bounds__(256) void tb_gemm(const __hip_bfloat16* __restrict__ A,
                                               const __hip_bfloat16* __restrict__ W,
                                               void* __restrict__ outp,
                                               const float* __restrict__ bias, int mode) {
  const int Kd = 1024;
  int tid = threadIdx.x;
  int wave = tid >> 5, lane = tid & 31, r = lane & 15, half = lane >> 4;
  int wm = wave >> 1, wn = wave & 1;                     // 4x2 wave grid
  int m0 = (int)(blockIdx.x >> 3) * 128 + wm * 32;       // 64 m-blocks
  int n0 = (int)(blockIdx.x & 7) * 128 + wn * 64;        // 8 n-blocks

  v8f acc[2][4];
  v8f z = {0.f, 0.f, 0.f, 0.f, 0.f, 0.f, 0.f, 0.f};
#pragma unroll
  for (int i = 0; i < 2; ++i)
#pragma unroll
    for (int j = 0; j < 4; ++j) acc[i][j] = z;

  for (int k0 = 0; k0 < Kd; k0 += 32) {
    if (k0 + 32 < Kd) {  // uniform branch; speculative prefetch of next A chunk
      __builtin_prefetch((const char*)(A + (size_t)(m0 + r) * Kd + k0 + 32), 0, 3);
    }
    v16bf a0 = load_a_g(A, Kd, m0, k0);
    v16bf a1 = load_a_g(A, Kd, m0 + 16, k0);
    v16bf b[4];
#pragma unroll
    for (int j = 0; j < 4; ++j) b[j] = load_b_g(W, Kd, n0 + j * 16, k0);
#pragma unroll
    for (int j = 0; j < 4; ++j) {
      acc[0][j] = wmma_bf16(a0, b[j], acc[0][j]);
      acc[1][j] = wmma_bf16(a1, b[j], acc[1][j]);
    }
  }

#pragma unroll
  for (int im = 0; im < 2; ++im)
#pragma unroll
    for (int jn = 0; jn < 4; ++jn) {
      int col = n0 + jn * 16 + r;
      int rowb = m0 + im * 16 + half * 8;
#pragma unroll
      for (int v = 0; v < 8; ++v) {
        int row = rowb + v;
        float val = acc[im][jn][v];
        if (mode <= 1) {
          // head-split: channel c -> (dh, h) with c = dh*H + h   (H=16, DH=64)
          int h = col & 15, dh = col >> 4;
          int b_ = row >> 10, n_ = row & 1023;
          __hip_bfloat16* dst = (__hip_bfloat16*)outp;
          dst[((((size_t)b_ * 16 + h) * 1024) + n_) * 64 + dh] = __float2bfloat16(val);
        } else if (mode == 2) {
          ((float*)outp)[(size_t)row * 1024 + col] = val;
        } else {
          ((float*)outp)[(size_t)row * 1024 + col] = val + bias[col];
        }
      }
    }
}

// ---------------------------------------------------------------- attention
// One workgroup per (b,h).  Stage K[1024x64] bf16 in LDS via the Tensor Data
// Mover (hardware-padded to 144B row stride -> conflict-free ds_load_b128),
// compute transposed score tiles C[key, query] = K . Q^T via WMMA; online
// softmax stats are lane-local; second sweep accumulates softmax column sums
// S[n] with LDS float atomics.
__global__ __launch_bounds__(256) void tb_attn(const __hip_bfloat16* __restrict__ Qh,
                                               const __hip_bfloat16* __restrict__ Kh,
                                               float* __restrict__ Ssum) {
  extern __shared__ char smem[];
  const int KSTR = 144;  // 64*2 payload + 16 pad (inserted by TDM), 16B aligned
  char* kLds = smem;
  float* S = (float*)(smem + 1024 * KSTR);

  int bh = blockIdx.x;  // 0..127 = b*16 + h
  const __hip_bfloat16* kg = Kh + (size_t)bh * 1024 * 64;
  const __hip_bfloat16* qg = Qh + (size_t)bh * 1024 * 64;
  int t = threadIdx.x;

  // ---- TDM: DMA K[1024x64] bf16 global -> LDS with padding.  One wave issues
  // the descriptor (tensor ops ignore EXEC); TENSORcnt is per-wave so the
  // issuing wave waits, then the workgroup barrier publishes the tile.
  if (t < 32) {
    uint64_t ga = (uint64_t)(uintptr_t)kg;
    v4u g0;
    g0.x = 1u;                                // count=1 (valid), is_restore=0
    g0.y = 0u;                                // lds_addr = 0 (smem base)
    g0.z = (unsigned)ga;                      // global_addr[31:0]
    g0.w = (unsigned)(ga >> 32) | (2u << 30); // global_addr[56:32], type=2
    v8u g1;
    g1.s0 = (1u << 16)      // data_size = 1 -> 2 bytes (bf16)
          | (1u << 20)      // pad_enable
          | (4u << 22)      // pad_interval: every 32 DWORDs (128B payload)
          | (3u << 25);     // pad_amount: 4 DWORDs (16B)
    g1.s1 = 64u << 16;      // tensor_dim0 = 64 elems (low 16 bits)
    g1.s2 = 1024u << 16;    // tensor_dim0 hi = 0 | tensor_dim1 = 1024 (low 16)
    g1.s3 = 64u << 16;      // tensor_dim1 hi = 0 | tile_dim0 = 64
    g1.s4 = 1024u;          // tile_dim1 = 1024, tile_dim2 = 0
    g1.s5 = 64u;            // tensor_dim0_stride = 64 elems (low 32)
    g1.s6 = 0u;             // stride hi, tensor_dim1_stride lo
    g1.s7 = 0u;
    asm volatile("tensor_load_to_lds %0, %1" :: "s"(g0), "s"(g1) : "memory");
    __builtin_amdgcn_s_wait_tensorcnt(0);
  }
  for (int i = t; i < 1024; i += 256) S[i] = 0.f;
  __syncthreads();

  int wave = t >> 5, lane = t & 31, r = lane & 15, half = lane >> 4;
  const float scale = 0.125f;  // 64^-0.5

  for (int blk = 0; blk < 8; ++blk) {
    int m0 = (wave + blk * 8) * 16;  // 16 query rows per wave-block
    // Q^T as B-matrix fragments (columns = queries), kept in regs both sweeps
    v16bf bq0 = load_b_g(qg, 64, m0, 0);
    v16bf bq1 = load_b_g(qg, 64, m0, 32);

    // ---- sweep 1: online row-max / row-sum; each lane covers 512 keys of query m0+r
    float M = -3.0e38f, L = 0.f;
    for (int nt = 0; nt < 64; ++nt) {
      const char* kt = kLds + (size_t)nt * 16 * KSTR;
      v16bf a0 = load_a_lds(kt, KSTR, 0);
      v16bf a1 = load_a_lds(kt, KSTR, 32);
      v8f c = {0.f, 0.f, 0.f, 0.f, 0.f, 0.f, 0.f, 0.f};
      c = wmma_bf16(a0, bq0, c);
      c = wmma_bf16(a1, bq1, c);
      float tmax = -3.0e38f;
#pragma unroll
      for (int v = 0; v < 8; ++v) tmax = fmaxf(tmax, c[v] * scale);
      float newM = fmaxf(M, tmax);
      float corr = __expf(M - newM);
      float s = 0.f;
#pragma unroll
      for (int v = 0; v < 8; ++v) s += __expf(c[v] * scale - newM);
      L = L * corr + s;
      M = newM;
    }
    // merge the two half-wave partials for query m0+r
    float Mo = __shfl_xor(M, 16);
    float Lo = __shfl_xor(L, 16);
    float Mf = fmaxf(M, Mo);
    float Lf = L * __expf(M - Mf) + Lo * __expf(Mo - Mf);
    float invL = 1.0f / Lf;

    // ---- sweep 2: recompute scores, accumulate column sums S[n] += sum_m p[m,n]
    for (int nt = 0; nt < 64; ++nt) {
      const char* kt = kLds + (size_t)nt * 16 * KSTR;
      v16bf a0 = load_a_lds(kt, KSTR, 0);
      v16bf a1 = load_a_lds(kt, KSTR, 32);
      v8f c = {0.f, 0.f, 0.f, 0.f, 0.f, 0.f, 0.f, 0.f};
      c = wmma_bf16(a0, bq0, c);
      c = wmma_bf16(a1, bq1, c);
#pragma unroll
      for (int v = 0; v < 8; ++v) {
        float p = __expf(c[v] * scale - Mf) * invL;
        // reduce over the 16 queries held in this half-wave (xor stays in-half)
        p += __shfl_xor(p, 1);
        p += __shfl_xor(p, 2);
        p += __shfl_xor(p, 4);
        p += __shfl_xor(p, 8);
        if (r == 0) atomicAdd(&S[nt * 16 + half * 8 + v], p);
      }
    }
  }
  __syncthreads();
  for (int i = t; i < 1024; i += 256) Ssum[(size_t)bh * 1024 + i] = S[i];
}

// ---------------------------------------------------------------- v * S, permute to merge-heads order, -> bf16
// y[row, h*64+d] = vraw[row, d*16+h] * S[b*16+h, n]   (row = b*1024+n)
__global__ __launch_bounds__(256) void tb_scale(const float* __restrict__ vraw,
                                                const float* __restrict__ Ssum,
                                                __hip_bfloat16* __restrict__ y) {
  size_t i = (size_t)blockIdx.x * 256 + threadIdx.x;  // 8M elements
  int c = (int)(i & 1023);
  size_t row = i >> 10;
  int h = c >> 6, d = c & 63;
  int b = (int)(row >> 10), n = (int)(row & 1023);
  float s = Ssum[(((size_t)b * 16 + h) << 10) + n];
  float val = vraw[(row << 10) + (size_t)(d * 16 + h)] * s;
  y[i] = __float2bfloat16(val);
}

// ---------------------------------------------------------------- launcher
extern "C" void kernel_launch(void* const* d_in, const int* in_sizes, int n_in,
                              void* d_out, int out_size, void* d_ws, size_t ws_size,
                              hipStream_t stream) {
  (void)in_sizes; (void)n_in; (void)out_size; (void)ws_size;
  const float* x  = (const float*)d_in[0];
  const float* Wq = (const float*)d_in[1];
  const float* Wk = (const float*)d_in[2];
  const float* Wv = (const float*)d_in[3];
  const float* Wo = (const float*)d_in[4];
  const float* bo = (const float*)d_in[5];

  char* ws = (char*)d_ws;
  const size_t XB = 16777216;  // 8192*1024 bf16
  const size_t WB = 2097152;   // 1024*1024 bf16
  __hip_bfloat16* xb  = (__hip_bfloat16*)(ws);
  __hip_bfloat16* wqb = (__hip_bfloat16*)(ws + XB);
  __hip_bfloat16* wkb = (__hip_bfloat16*)(ws + XB + WB);
  __hip_bfloat16* wvb = (__hip_bfloat16*)(ws + XB + 2 * WB);
  __hip_bfloat16* wob = (__hip_bfloat16*)(ws + XB + 3 * WB);
  __hip_bfloat16* qh  = (__hip_bfloat16*)(ws + XB + 4 * WB);            // bf16 [b,h,n,dh]
  __hip_bfloat16* kh  = (__hip_bfloat16*)(ws + 2 * XB + 4 * WB);        // bf16 [b,h,n,dh]
  float* vraw         = (float*)(ws + 3 * XB + 4 * WB);                  // f32 [8192,1024]
  float* Ssum         = (float*)(ws + 3 * XB + 4 * WB + 33554432);       // f32 [128,1024]
  __hip_bfloat16* yb  = (__hip_bfloat16*)(ws + 3 * XB + 4 * WB + 33554432 + 524288);

  // fp32 -> bf16
  tb_cvt_bf16<<<(8388608 + 255) / 256, 256, 0, stream>>>(x, xb, 8388608);
  tb_cvt_bf16<<<(1048576 + 255) / 256, 256, 0, stream>>>(Wq, wqb, 1048576);
  tb_cvt_bf16<<<(1048576 + 255) / 256, 256, 0, stream>>>(Wk, wkb, 1048576);
  tb_cvt_bf16<<<(1048576 + 255) / 256, 256, 0, stream>>>(Wv, wvb, 1048576);
  tb_cvt_bf16<<<(1048576 + 255) / 256, 256, 0, stream>>>(Wo, wob, 1048576);

  // QKV projections (512 blocks = 64 m-tiles x 8 n-tiles, 256 thr = 8 waves)
  tb_gemm<<<512, 256, 0, stream>>>(xb, wqb, (void*)qh, nullptr, 0);
  tb_gemm<<<512, 256, 0, stream>>>(xb, wkb, (void*)kh, nullptr, 1);
  tb_gemm<<<512, 256, 0, stream>>>(xb, wvb, (void*)vraw, nullptr, 2);

  // attention column sums: 1 WG per (b,h); 1024*144 + 4096 = 151552B dynamic LDS
  const int attn_lds = 1024 * 144 + 4096;
  (void)hipFuncSetAttribute((const void*)tb_attn,
                            hipFuncAttributeMaxDynamicSharedMemorySize, attn_lds);
  tb_attn<<<128, 256, attn_lds, stream>>>(qh, kh, Ssum);

  // v * colsum(softmax), merge heads, to bf16
  tb_scale<<<32768, 256, 0, stream>>>(vraw, Ssum, yb);

  // output projection + bias -> fp32 d_out
  tb_gemm<<<512, 256, 0, stream>>>(yb, wob, d_out, bo, 3);
}
